// GatedCrossAttention_77678778515532
// MI455X (gfx1250) — compile-verified, matrix operand order checked by reading, not run on hardware
//
#include <hip/hip_runtime.h>
#include <hip/hip_bf16.h>
#include <math.h>
#include <stdint.h>

// Problem constants (match reference)
#define L_DIM 2048
#define C_DIM 2048
#define B_DIM 4
#define D_DIM 1024
#define Z_DIM 128
#define ZZ    256            // augmented 2Z (q|rotary_a , k|rotary_b)
#define E_DIM 2176           // 2D + Z
#define MROWS 8192           // L*B == C*B

typedef __attribute__((ext_vector_type(16))) __bf16 v16bf;
typedef __attribute__((ext_vector_type(8)))  __bf16 bf16x8;   // 16-byte chunk
typedef __attribute__((ext_vector_type(8)))  float  v8f;

#if __has_builtin(__builtin_amdgcn_tensor_load_to_lds)
#define USE_TDM 1
#else
#define USE_TDM 0
#endif

__device__ __forceinline__ float sigmoidf_(float x) { return 1.0f / (1.0f + __expf(-x)); }
__device__ __forceinline__ float siluf_(float x)    { return x * sigmoidf_(x); }

// CDNA5 16-bit WMMA fragment layout (ISA 7.12.2):
//   lane l holds row (l&15); element e maps to K = (e&7) + 8*(l>>4) + 16*(e>>3)
// => per-lane two contiguous 16B chunks at K and K+16.  Works for A (MxK row-major)
// and for B given as (N x K) row-major (i.e. weights W[N][K], or K' rows, or V^T rows).
__device__ __forceinline__ v16bf load_frag(const __bf16* base, int ld, int row, int kbase, int lane)
{
    const __bf16* p = base + (size_t)(row + (lane & 15)) * ld + (kbase + ((lane >> 4) << 3));
    bf16x8 lo = *(const bf16x8*)p;
    bf16x8 hi = *(const bf16x8*)(p + 16);
    v16bf r;
#pragma unroll
    for (int i = 0; i < 8; ++i) { r[i] = lo[i]; r[i + 8] = hi[i]; }
    return r;
}

__device__ __forceinline__ v8f wmma_bf16(v16bf a, v16bf b, v8f c)
{
    return __builtin_amdgcn_wmma_f32_16x16x32_bf16(false, a, false, b, (short)0, c, false, false);
}

#if USE_TDM
typedef __attribute__((ext_vector_type(4))) unsigned int u32x4;
typedef __attribute__((ext_vector_type(8))) int          i32x8;
typedef __attribute__((ext_vector_type(4))) int          i32x4;

// Issue one TDM descriptor: 2D tile (rows x 32 bf16) from row-major global (ld = K
// elements) into LDS with a padded row stride of 40 bf16 (pad 4 DWORDs per 16 DWORDs).
__device__ __forceinline__ void tdm_load_tile(const __bf16* gsrc, unsigned lds_off,
                                              int K, int rows)
{
    unsigned long long ga = (unsigned long long)(uintptr_t)gsrc;
    u32x4 g0;
    g0[0] = 1u;                                                 // count=1, user descriptor
    g0[1] = lds_off;                                            // lds_addr
    g0[2] = (unsigned)ga;                                       // global_addr[31:0]
    g0[3] = (unsigned)((ga >> 32) & 0x01FFFFFFu) | (2u << 30);  // global_addr[56:32] | type=2
    i32x8 g1;
    g1[0] = (1 << 16)      // data_size = 2 bytes
          | (1 << 20)      // pad_enable
          | (3 << 22)      // pad_interval: every 16 DWORDs (one 32-elem bf16 line)
          | (3 << 25);     // pad_amount: 4 DWORDs (8 bf16) -> LDS line stride 40
    g1[1] = (K & 0xffff) << 16;                                 // tensor_dim0[15:0]
    g1[2] = ((K >> 16) & 0xffff) | ((rows & 0xffff) << 16);     // td0[31:16] | td1[15:0]
    g1[3] = (32 << 16);                                         // td1[31:16]=0 | tile_dim0=32
    g1[4] = (rows & 0xffff);                                    // tile_dim1 | tile_dim2=0
    g1[5] = K;                                                  // tensor_dim0_stride[31:0]
    g1[6] = 0;
    g1[7] = 0;
    i32x4 gz; gz[0] = 0; gz[1] = 0; gz[2] = 0; gz[3] = 0;
#if defined(__clang_major__) && (__clang_major__ >= 23)
    i32x8 gz8; 
#pragma unroll
    for (int i = 0; i < 8; ++i) gz8[i] = 0;
    __builtin_amdgcn_tensor_load_to_lds(g0, g1, gz, gz, gz8, 0);
#else
    __builtin_amdgcn_tensor_load_to_lds(g0, g1, gz, gz, 0);
#endif
}
#endif

// ---------------------------------------------------------------- utilities
__global__ void cvt_f32_bf16(const float* __restrict__ in, __bf16* __restrict__ out, int n)
{
    int i = blockIdx.x * blockDim.x + threadIdx.x;
    int stride = gridDim.x * blockDim.x;
    for (; i < n; i += stride) out[i] = (__bf16)in[i];
}

// LayerNorm over D=1024 per (l,b) row; emits bf16 normalized x
__global__ __launch_bounds__(128) void ln_kernel(const float* __restrict__ q,
                                                 const float* __restrict__ w,
                                                 const float* __restrict__ bvec,
                                                 __bf16* __restrict__ xout)
{
    __shared__ float red[128];
    const int row = blockIdx.x, tid = threadIdx.x;
    const float* p = q + (size_t)row * D_DIM;
    float vals[8], s = 0.f;
#pragma unroll
    for (int j = 0; j < 8; ++j) { vals[j] = p[tid + j * 128]; s += vals[j]; }
    red[tid] = s; __syncthreads();
    for (int off = 64; off > 0; off >>= 1) { if (tid < off) red[tid] += red[tid + off]; __syncthreads(); }
    float mu = red[0] * (1.0f / D_DIM);
    __syncthreads();
    float vs = 0.f;
#pragma unroll
    for (int j = 0; j < 8; ++j) { float d = vals[j] - mu; vs += d * d; }
    red[tid] = vs; __syncthreads();
    for (int off = 64; off > 0; off >>= 1) { if (tid < off) red[tid] += red[tid + off]; __syncthreads(); }
    float inv = rsqrtf(red[0] * (1.0f / D_DIM) + 1e-5f);
    __bf16* o = xout + (size_t)row * D_DIM;
#pragma unroll
    for (int j = 0; j < 8; ++j) {
        int e = tid + j * 128;
        o[e] = (__bf16)((vals[j] - mu) * inv * w[e] + bvec[e]);
    }
}

// Rotary rank-Z bias, folded as extra Z columns of Q'/K' (bias = a @ b^T)
__global__ __launch_bounds__(64) void rotary_kernel(const float* __restrict__ alpha,
                                                    const float* __restrict__ beta,
                                                    __bf16* __restrict__ qbuf,
                                                    __bf16* __restrict__ kbuf)
{
    const int n = blockIdx.x;           // position 0..2047 (L==C)
    const int i = threadIdx.x;          // 0..63 (Z/2)
    const float f = __expf(-(float)i * 0.14391156831212787f); // ln(1e4)/64
    float ang = (float)n * f;
    float sn = __sinf(ang), cs = __cosf(ang);
    float a_lo = alpha[i] * cs - alpha[i + 64] * sn;
    float a_hi = alpha[i + 64] * cs + alpha[i] * sn;
    float b_lo = beta[i]  * cs - beta[i + 64]  * sn;
    float b_hi = beta[i + 64]  * cs + beta[i]  * sn;
#pragma unroll
    for (int bb = 0; bb < B_DIM; ++bb) {
        size_t o = ((size_t)bb * L_DIM + n) * ZZ + Z_DIM;
        qbuf[o + i]      = (__bf16)a_lo;
        qbuf[o + 64 + i] = (__bf16)a_hi;
        kbuf[o + i]      = (__bf16)b_lo;
        kbuf[o + 64 + i] = (__bf16)b_hi;
    }
}

// ---------------------------------------------------------------- GEMM
// C[M,N] = A[M,K] @ W[N,K]^T + bias, bf16 inputs, f32 accum, fused epilogue.
// Block tile 128x128, BK=32, 8 waves (2x4), wave tile 64x32, double-buffered LDS.
// Tile staging done by the Tensor Data Mover (one wave issues descriptors; DMA fills
// LDS incl. the 40-element padded row stride) so all waves only do ds_load + WMMA.
// MODE 0: base proj  -> u=sigmoid (f32), r=silu (bf16), q*Z^-0.5 -> Q' buffer
// MODE 1: k proj     -> K' buffer (per-batch row-major C x 2Z, first Z cols)
// MODE 2: v proj     -> silu, stored TRANSPOSED per batch (D x C) for attn B-operand
// MODE 3: out proj   -> silu + final gate: out = res + u*(h - res)
template<int MODE>
__global__ __launch_bounds__(256)
void gemm_bf16_kernel(const __bf16* __restrict__ A, const __bf16* __restrict__ Bw,
                      const float* __restrict__ bias, int K,
                      float* __restrict__ fout, __bf16* __restrict__ bout,
                      __bf16* __restrict__ bout2, const float* __restrict__ resid,
                      const float* __restrict__ uin)
{
    __shared__ __bf16 As[2][128 * 40];
    __shared__ __bf16 Bs[2][128 * 40];
    const int tid = threadIdx.x, lane = tid & 31, wave = tid >> 5;
    const int wm = (wave >> 2) * 64, wn = (wave & 3) * 32;
    const int m0 = blockIdx.y * 128, n0 = blockIdx.x * 128;
    const __bf16* Ag = A  + (size_t)m0 * K;
    const __bf16* Bg = Bw + (size_t)n0 * K;

    v8f acc[4][2];
#pragma unroll
    for (int i = 0; i < 4; ++i)
#pragma unroll
        for (int j = 0; j < 2; ++j)
#pragma unroll
            for (int e = 0; e < 8; ++e) acc[i][j][e] = 0.0f;

    int buf = 0;
#if USE_TDM
    if (wave == 0) {
        tdm_load_tile(Ag, (unsigned)(uintptr_t)&As[0][0], K, 128);
        tdm_load_tile(Bg, (unsigned)(uintptr_t)&Bs[0][0], K, 128);
    }
    for (int kk = 0; kk < K; kk += 32) {
        if (wave == 0) __builtin_amdgcn_s_wait_tensorcnt((short)0);  // current tile DMA done
        __syncthreads();                                             // publish to all waves
        if (wave == 0 && (kk + 32) < K) {                            // prefetch next tile
            tdm_load_tile(Ag + (kk + 32), (unsigned)(uintptr_t)&As[buf ^ 1][0], K, 128);
            tdm_load_tile(Bg + (kk + 32), (unsigned)(uintptr_t)&Bs[buf ^ 1][0], K, 128);
        }
        v16bf af[4], bfg[2];
#pragma unroll
        for (int i = 0; i < 4; ++i) af[i]  = load_frag(&As[buf][0], 40, wm + i * 16, 0, lane);
#pragma unroll
        for (int j = 0; j < 2; ++j) bfg[j] = load_frag(&Bs[buf][0], 40, wn + j * 16, 0, lane);
#pragma unroll
        for (int i = 0; i < 4; ++i)
#pragma unroll
            for (int j = 0; j < 2; ++j)
                acc[i][j] = wmma_bf16(af[i], bfg[j], acc[i][j]);
        buf ^= 1;
    }
#else
    // Fallback: register-staged global->LDS copies (round-1 path)
    const int c1 = tid + 256;
    const int ar0 = tid >> 2, ak0 = (tid & 3) * 8;
    const int ar1 = c1 >> 2,  ak1 = (c1 & 3) * 8;
    bf16x8 ra0 = *(const bf16x8*)&Ag[(size_t)ar0 * K + ak0];
    bf16x8 ra1 = *(const bf16x8*)&Ag[(size_t)ar1 * K + ak1];
    bf16x8 rb0 = *(const bf16x8*)&Bg[(size_t)ar0 * K + ak0];
    bf16x8 rb1 = *(const bf16x8*)&Bg[(size_t)ar1 * K + ak1];
    *(bf16x8*)&As[0][ar0 * 40 + ak0] = ra0;
    *(bf16x8*)&As[0][ar1 * 40 + ak1] = ra1;
    *(bf16x8*)&Bs[0][ar0 * 40 + ak0] = rb0;
    *(bf16x8*)&Bs[0][ar1 * 40 + ak1] = rb1;
    __syncthreads();
    for (int kk = 0; kk < K; kk += 32) {
        const bool hasNext = (kk + 32) < K;
        if (hasNext) {
            const int kn = kk + 32;
            ra0 = *(const bf16x8*)&Ag[(size_t)ar0 * K + kn + ak0];
            ra1 = *(const bf16x8*)&Ag[(size_t)ar1 * K + kn + ak1];
            rb0 = *(const bf16x8*)&Bg[(size_t)ar0 * K + kn + ak0];
            rb1 = *(const bf16x8*)&Bg[(size_t)ar1 * K + kn + ak1];
        }
        v16bf af[4], bfg[2];
#pragma unroll
        for (int i = 0; i < 4; ++i) af[i]  = load_frag(&As[buf][0], 40, wm + i * 16, 0, lane);
#pragma unroll
        for (int j = 0; j < 2; ++j) bfg[j] = load_frag(&Bs[buf][0], 40, wn + j * 16, 0, lane);
#pragma unroll
        for (int i = 0; i < 4; ++i)
#pragma unroll
            for (int j = 0; j < 2; ++j)
                acc[i][j] = wmma_bf16(af[i], bfg[j], acc[i][j]);
        if (hasNext) {
            const int nb = buf ^ 1;
            *(bf16x8*)&As[nb][ar0 * 40 + ak0] = ra0;
            *(bf16x8*)&As[nb][ar1 * 40 + ak1] = ra1;
            *(bf16x8*)&Bs[nb][ar0 * 40 + ak0] = rb0;
            *(bf16x8*)&Bs[nb][ar1 * 40 + ak1] = rb1;
        }
        __syncthreads();
        buf ^= 1;
    }
#endif

    // Epilogue. C/D layout: n = lane&15, m = e + 8*(lane>>4).
#pragma unroll
    for (int i = 0; i < 4; ++i) {
#pragma unroll
        for (int j = 0; j < 2; ++j) {
            const int ng = n0 + wn + j * 16 + (lane & 15);
            const float bsc = bias[ng];
#pragma unroll
            for (int e = 0; e < 8; ++e) {
                const int mg = m0 + wm + i * 16 + ((lane >> 4) << 3) + e;
                const float val = acc[i][j][e] + bsc;
                if constexpr (MODE == 0) {
                    if (ng < D_DIM) {
                        fout[(size_t)mg * D_DIM + ng] = sigmoidf_(val);              // u
                    } else if (ng < 2 * D_DIM) {
                        bout[(size_t)mg * D_DIM + (ng - D_DIM)] = (__bf16)siluf_(val); // r
                    } else {
                        const int bb = mg & 3, ll = mg >> 2;
                        bout2[((size_t)bb * L_DIM + ll) * ZZ + (ng - 2 * D_DIM)] =
                            (__bf16)(val * 0.08838834764831845f);                    // q * Z^-0.5
                    }
                } else if constexpr (MODE == 1) {
                    const int bb = mg & 3, cc = mg >> 2;
                    bout[((size_t)bb * C_DIM + cc) * ZZ + ng] = (__bf16)val;         // k
                } else if constexpr (MODE == 2) {
                    const int bb = mg & 3, cc = mg >> 2;
                    bout[((size_t)bb * D_DIM + ng) * C_DIM + cc] = (__bf16)siluf_(val); // V^T
                } else {
                    const float hs = siluf_(val);
                    const size_t idx = (size_t)mg * D_DIM + ng;
                    const float res = resid[idx];
                    fout[idx] = res + uin[idx] * (hs - res);                          // final gate
                }
            }
        }
    }
}

// ---------------------------------------------------------------- flash attention
// Grid (L/16, B). 16 query rows per WG; key chunks of 64; waves 0-3 build S via
// WMMA (Q' 16x256 x K'^T), online softmax in LDS, all 8 waves do P@V^T, each wave
// owning 128 of the D=1024 output columns (8x v8f accumulators).
__global__ __launch_bounds__(256)
void flash_attn_kernel(const __bf16* __restrict__ qbuf, const __bf16* __restrict__ kbuf,
                       const __bf16* __restrict__ vtbuf, const __bf16* __restrict__ rbuf,
                       __bf16* __restrict__ hrbuf)
{
    __shared__ float  sS[16][68];
    __shared__ __bf16 sP[16][72];
    __shared__ float  sM[16], sL[16], sA[16];

    const int b = blockIdx.y;
    const int l0 = blockIdx.x * 16;
    const int tid = threadIdx.x, lane = tid & 31, wave = tid >> 5;

    const __bf16* Qb = qbuf  + (size_t)b * L_DIM * ZZ;
    const __bf16* Kb = kbuf  + (size_t)b * C_DIM * ZZ;
    const __bf16* Vt = vtbuf + (size_t)b * D_DIM * C_DIM;

    v16bf qf[8];
    if (wave < 4) {
#pragma unroll
        for (int ks = 0; ks < 8; ++ks) qf[ks] = load_frag(Qb, ZZ, l0, ks * 32, lane);
    }
    v8f o[8];
#pragma unroll
    for (int j = 0; j < 8; ++j)
#pragma unroll
        for (int e = 0; e < 8; ++e) o[j][e] = 0.0f;
    if (tid < 16) { sM[tid] = -3.0e38f; sL[tid] = 0.0f; }
    __syncthreads();

    for (int c0 = 0; c0 < C_DIM; c0 += 64) {
        if (wave < 4) {                        // S = Q' @ K'^T  (16 x 64 slab)
            v8f s;
#pragma unroll
            for (int e = 0; e < 8; ++e) s[e] = 0.0f;
#pragma unroll
            for (int ks = 0; ks < 8; ++ks) {
                v16bf kf = load_frag(Kb, ZZ, c0 + wave * 16, ks * 32, lane);
                s = wmma_bf16(qf[ks], kf, s);
            }
            const int col = wave * 16 + (lane & 15);
#pragma unroll
            for (int e = 0; e < 8; ++e) sS[e + ((lane >> 4) << 3)][col] = s[e];
        }
        if (c0 + 64 < C_DIM) {                 // hint next V chunk toward the caches
            __builtin_prefetch(Vt + (size_t)(wave * 128 + (lane & 15)) * C_DIM + c0 + 64, 0, 1);
        }
        __syncthreads();
        if (tid < 16) {                        // online softmax, one row per thread
            float mc = -3.0e38f;
#pragma unroll 4
            for (int j = 0; j < 64; ++j) mc = fmaxf(mc, sS[tid][j]);
            const float mold = sM[tid];
            const float mnew = fmaxf(mold, mc);
            const float al = __expf(mold - mnew);
            float lsum = 0.0f;
#pragma unroll 4
            for (int j = 0; j < 64; ++j) {
                const float p = __expf(sS[tid][j] - mnew);
                lsum += p;
                sP[tid][j] = (__bf16)p;
            }
            sM[tid] = mnew; sL[tid] = sL[tid] * al + lsum; sA[tid] = al;
        }
        __syncthreads();
        float av[8];
#pragma unroll
        for (int e = 0; e < 8; ++e) av[e] = sA[e + ((lane >> 4) << 3)];
        v16bf pf[2];
#pragma unroll
        for (int ks = 0; ks < 2; ++ks) pf[ks] = load_frag(&sP[0][0], 72, 0, ks * 32, lane);
#pragma unroll
        for (int j = 0; j < 8; ++j) {
#pragma unroll
            for (int e = 0; e < 8; ++e) o[j][e] *= av[e];
            const int ecol = wave * 128 + j * 16;
#pragma unroll
            for (int ks = 0; ks < 2; ++ks) {
                v16bf vf = load_frag(Vt, C_DIM, ecol, c0 + ks * 32, lane);
                o[j] = wmma_bf16(pf[ks], vf, o[j]);
            }
        }
        __syncthreads();
    }

    float inv[8];
#pragma unroll
    for (int e = 0; e < 8; ++e) inv[e] = 1.0f / sL[e + ((lane >> 4) << 3)];
#pragma unroll
    for (int j = 0; j < 8; ++j) {
#pragma unroll
        for (int e = 0; e < 8; ++e) {
            const int lg = l0 + ((lane >> 4) << 3) + e;
            const int dg = wave * 128 + j * 16 + (lane & 15);
            const size_t idx = ((size_t)lg * B_DIM + b) * D_DIM + dg;
            const float h = o[j][e] * inv[e];
            hrbuf[idx] = (__bf16)(h * (float)rbuf[idx]);   // fuse h * r for wh GEMM
        }
    }
}

// ---------------------------------------------------------------- launch
extern "C" void kernel_launch(void* const* d_in, const int* in_sizes, int n_in,
                              void* d_out, int out_size, void* d_ws, size_t ws_size,
                              hipStream_t stream)
{
    (void)in_sizes; (void)n_in; (void)out_size; (void)ws_size;
    const float* query  = (const float*)d_in[0];
    const float* keyseq = (const float*)d_in[1];
    const float* value  = (const float*)d_in[2];
    const float* wq = (const float*)d_in[3];
    const float* bq = (const float*)d_in[4];
    const float* wk = (const float*)d_in[5];
    const float* bk = (const float*)d_in[6];
    const float* wv = (const float*)d_in[7];
    const float* bv = (const float*)d_in[8];
    const float* wh = (const float*)d_in[9];
    const float* bh = (const float*)d_in[10];
    const float* lnw = (const float*)d_in[11];
    const float* lnb = (const float*)d_in[12];
    const float* alpha = (const float*)d_in[13];
    const float* beta  = (const float*)d_in[14];
    float* out = (float*)d_out;

    char* ws = (char*)d_ws;
    size_t off = 0;
    auto alloc = [&](size_t bytes) -> char* {
        char* p = ws + off; off += (bytes + 255) & ~(size_t)255; return p;
    };
    __bf16* xb    = (__bf16*)alloc((size_t)MROWS * D_DIM * 2);  // LN(query) bf16
    __bf16* keyb  = (__bf16*)alloc((size_t)MROWS * D_DIM * 2);
    __bf16* valb  = (__bf16*)alloc((size_t)MROWS * D_DIM * 2);
    __bf16* wqb   = (__bf16*)alloc((size_t)E_DIM * D_DIM * 2);
    __bf16* wkb   = (__bf16*)alloc((size_t)Z_DIM * D_DIM * 2);
    __bf16* wvb   = (__bf16*)alloc((size_t)D_DIM * D_DIM * 2);
    __bf16* whb   = (__bf16*)alloc((size_t)D_DIM * D_DIM * 2);
    float*  ubuf  = (float*) alloc((size_t)MROWS * D_DIM * 4);
    __bf16* rbuf  = (__bf16*)alloc((size_t)MROWS * D_DIM * 2);
    __bf16* qbuf  = (__bf16*)alloc((size_t)B_DIM * L_DIM * ZZ * 2); // Q' = [q*s | a]
    __bf16* kbuf  = (__bf16*)alloc((size_t)B_DIM * C_DIM * ZZ * 2); // K' = [k | b]
    __bf16* vtbuf = (__bf16*)alloc((size_t)B_DIM * D_DIM * C_DIM * 2); // V^T per batch
    __bf16* hrbuf = (__bf16*)alloc((size_t)MROWS * D_DIM * 2);

    auto cvt = [&](const float* src, __bf16* dst, int n) {
        int blocks = (n + 2047) / 2048;
        cvt_f32_bf16<<<blocks, 256, 0, stream>>>(src, dst, n);
    };
    cvt(wq, wqb, E_DIM * D_DIM);
    cvt(wk, wkb, Z_DIM * D_DIM);
    cvt(wv, wvb, D_DIM * D_DIM);
    cvt(wh, whb, D_DIM * D_DIM);
    cvt(keyseq, keyb, MROWS * D_DIM);
    cvt(value,  valb, MROWS * D_DIM);

    ln_kernel<<<MROWS, 128, 0, stream>>>(query, lnw, lnb, xb);
    rotary_kernel<<<2048, 64, 0, stream>>>(alpha, beta, qbuf, kbuf);

    // base = LN(x) @ wq^T + bq  -> u / r / q
    gemm_bf16_kernel<0><<<dim3(E_DIM / 128, MROWS / 128), 256, 0, stream>>>(
        xb, wqb, bq, D_DIM, ubuf, rbuf, qbuf, nullptr, nullptr);
    // k = key @ wk^T + bk
    gemm_bf16_kernel<1><<<dim3(1, MROWS / 128), 256, 0, stream>>>(
        keyb, wkb, bk, D_DIM, nullptr, kbuf, nullptr, nullptr, nullptr);
    // v = silu(value @ wv^T + bv), stored transposed
    gemm_bf16_kernel<2><<<dim3(D_DIM / 128, MROWS / 128), 256, 0, stream>>>(
        valb, wvb, bv, D_DIM, nullptr, vtbuf, nullptr, nullptr, nullptr);
    // attention (bias folded into Q'/K'), emits h*r in bf16
    flash_attn_kernel<<<dim3(L_DIM / 16, B_DIM), 256, 0, stream>>>(
        qbuf, kbuf, vtbuf, rbuf, hrbuf);
    // out = res + u*(silu((h*r)@wh^T + bh) - res)
    gemm_bf16_kernel<3><<<dim3(D_DIM / 128, MROWS / 128), 256, 0, stream>>>(
        hrbuf, whb, bh, D_DIM, out, nullptr, nullptr, query, ubuf);
}
